// NeuronCircuit_70755291234742
// MI455X (gfx1250) — matrix operand order repaired and checked by reading, not compiled, not run on hardware
//
#include <hip/hip_runtime.h>
#include <hip/hip_bf16.h>
#include <math.h>

// ---------------------------------------------------------------------------
// Problem constants (match the reference)
// ---------------------------------------------------------------------------
#define kB   4
#define kS   1024
#define kD   1024
#define kR   64
#define kN   32
#define kH   16
#define kDH  64
#define kTOK (kB * kS)   // 4096 tokens
#define kNR  (kN * kR)   // 2048

typedef __attribute__((ext_vector_type(16))) __bf16 v16bf;
typedef __attribute__((ext_vector_type(8)))  __bf16 bf16x8;   // 16 bytes
typedef __attribute__((ext_vector_type(8)))  float  v8f;

union FragA { v16bf v; bf16x8 h[2]; };

// ---------------------------------------------------------------------------
// Elementwise / layout kernels
// ---------------------------------------------------------------------------
__global__ void f32_to_bf16_kernel(const float* __restrict__ in,
                                   __bf16* __restrict__ out, int n) {
  int i = blockIdx.x * 256 + threadIdx.x;
  if (i < n) out[i] = (__bf16)in[i];
}

// compress B: out[(n*64+r)*1024 + d] = neurons[n, d, r]   (neurons [N,D,R])
__global__ void compressB_kernel(const float* __restrict__ nr,
                                 __bf16* __restrict__ out) {
  int o = blockIdx.x * 256 + threadIdx.x;          // kNR*kD = 2M
  int n = o >> 16;
  int r = (o >> 10) & 63;
  int d = o & 1023;
  out[o] = (__bf16)nr[n * (kD * kR) + d * kR + r];
}

// restore B: out[d*2048 + c] = neuronsR[c, d]   (neuronsR flat [N*R, D])
__global__ void restoreB_kernel(const float* __restrict__ nr,
                                __bf16* __restrict__ out) {
  int o = blockIdx.x * 256 + threadIdx.x;          // kD*kNR = 2M
  int d = o >> 11;
  int c = o & 2047;
  out[o] = (__bf16)nr[(size_t)c * kD + d];
}

// h[bs,r] = sum_n w[bs,n] * xn[bs, n*64 + r]
__global__ void compress_reduce_kernel(const float* __restrict__ xn,
                                       const float* __restrict__ w,
                                       float* __restrict__ h) {
  int bs = blockIdx.x;
  int r  = threadIdx.x;                             // 64 threads
  const float* wp = w + (size_t)bs * kN;
  const float* xp = xn + (size_t)bs * kNR + r;
  float acc = 0.f;
#pragma unroll
  for (int n = 0; n < kN; ++n) acc += wp[n] * xp[n * kR];
  h[(size_t)bs * kR + r] = acc;
}

// z[bs, n*64+r] = bf16(w[bs,n] * h[bs,r])
__global__ void build_z_kernel(const float* __restrict__ w,
                               const float* __restrict__ h,
                               __bf16* __restrict__ z) {
  int bs = blockIdx.x;
  for (int c = threadIdx.x; c < kNR; c += 256) {
    int n = c >> 6, r = c & 63;
    z[(size_t)bs * kNR + c] = (__bf16)(w[(size_t)bs * kN + n] * h[(size_t)bs * kR + r]);
  }
}

// Vt[(b*H+h)*64 + dim][s] = V[(b*S+s)*D + h*64 + dim]
__global__ void transposeV_kernel(const __bf16* __restrict__ V,
                                  __bf16* __restrict__ Vt) {
  int o   = blockIdx.x * 256 + threadIdx.x;         // B*H*64*S = 4M
  int s   = o & 1023;
  int dim = (o >> 10) & 63;
  int bh  = o >> 16;
  int b   = bh >> 4, hh = bh & 15;
  Vt[o] = V[((size_t)(b * kS + s)) * kD + hh * kDH + dim];
}

// ---------------------------------------------------------------------------
// Tiled bf16 WMMA GEMM: C[M,N] = A[M,K] * Bt[N,K]^T   (Bt stored [N,K])
// Block: 256 threads = 8 waves (2x4), block tile 128x256, wave tile 64x64
// (4x4 WMMA grid -> 16 v_wmma per 32-wide K-step), f32 accumulate.
// ---------------------------------------------------------------------------
template <bool OUT_BF16>
__global__ __launch_bounds__(256) void gemm_bf16_kernel(
    const __bf16* __restrict__ A, const __bf16* __restrict__ Bt,
    void* __restrict__ C, int M, int Nn, int K) {
  __shared__ alignas(16) __bf16 As[128][40];
  __shared__ alignas(16) __bf16 Bs[256][40];

  const int tid  = threadIdx.x;
  const int lane = tid & 31, wave = tid >> 5;
  const int half = lane >> 4, l16 = lane & 15;
  const int wr = wave >> 2, wc = wave & 3;          // 2x4 wave grid
  const int bm = blockIdx.y * 128, bn = blockIdx.x * 256;

  v8f acc[4][4];
#pragma unroll
  for (int i = 0; i < 4; ++i)
#pragma unroll
    for (int j = 0; j < 4; ++j)
#pragma unroll
      for (int v = 0; v < 8; ++v) acc[i][j][v] = 0.f;

  const int ar = tid >> 1, ac = (tid & 1) * 16;     // A tile: 128 rows x 32
  const int br = tid;                               // B tile: 256 rows x 32

  for (int k0 = 0; k0 < K; k0 += 32) {
    const __bf16* ag = &A[(size_t)(bm + ar) * K + k0 + ac];
    const __bf16* bg = &Bt[(size_t)(bn + br) * K + k0];
    *(bf16x8*)&As[ar][ac]     = *(const bf16x8*)ag;
    *(bf16x8*)&As[ar][ac + 8] = *(const bf16x8*)(ag + 8);
    *(bf16x8*)&Bs[br][0]      = *(const bf16x8*)bg;
    *(bf16x8*)&Bs[br][8]      = *(const bf16x8*)(bg + 8);
    *(bf16x8*)&Bs[br][16]     = *(const bf16x8*)(bg + 16);
    *(bf16x8*)&Bs[br][24]     = *(const bf16x8*)(bg + 24);
    if (k0 + 32 < K) {                              // prefetch next K-tile
      __builtin_prefetch(ag + 32, 0, 3);
      __builtin_prefetch(bg + 32, 0, 3);
    }
    __syncthreads();

    FragA afr[4], bfr[4];
#pragma unroll
    for (int mt = 0; mt < 4; ++mt) {
      int row = wr * 64 + mt * 16 + l16;            // A: M = lane%16
      afr[mt].h[0] = *(const bf16x8*)&As[row][half * 8];        // K = half*8 + e
      afr[mt].h[1] = *(const bf16x8*)&As[row][16 + half * 8];   // K = 16+half*8+e
    }
#pragma unroll
    for (int nt = 0; nt < 4; ++nt) {
      int row = wc * 64 + nt * 16 + l16;            // B: N = lane%16
      bfr[nt].h[0] = *(const bf16x8*)&Bs[row][half * 8];        // K = half*8 + e
      bfr[nt].h[1] = *(const bf16x8*)&Bs[row][16 + half * 8];
    }
#pragma unroll
    for (int mt = 0; mt < 4; ++mt)
#pragma unroll
      for (int nt = 0; nt < 4; ++nt)
        acc[mt][nt] = __builtin_amdgcn_wmma_f32_16x16x32_bf16(
            false, afr[mt].v, false, bfr[nt].v, (short)0, acc[mt][nt], false, false);
    __syncthreads();
  }

#pragma unroll
  for (int mt = 0; mt < 4; ++mt)
#pragma unroll
    for (int nt = 0; nt < 4; ++nt)
#pragma unroll
      for (int v = 0; v < 8; ++v) {
        int row = bm + wr * 64 + mt * 16 + v + half * 8;
        int col = bn + wc * 64 + nt * 16 + l16;
        if (OUT_BF16)
          ((__bf16*)C)[(size_t)row * Nn + col] = (__bf16)acc[mt][nt][v];
        else
          ((float*)C)[(size_t)row * Nn + col] = acc[mt][nt][v];
      }
}

// ---------------------------------------------------------------------------
// Flash attention (causal), 16 heads of dh=64.
// Block: 128 threads = 4 waves; each wave handles 16 query rows.
// Per 32-key block: 2x2 WMMA for S = Q*K^T, online softmax, LDS round-trip
// for P re-layout (C->A fragment), 4 WMMA for O += P*V (V pre-transposed).
// Masking only on the diagonal block (wave-uniform branch).
// ---------------------------------------------------------------------------
__global__ __launch_bounds__(128) void flash_attn_kernel(
    const __bf16* __restrict__ Q, const __bf16* __restrict__ Km,
    const __bf16* __restrict__ Vt, __bf16* __restrict__ ctx) {
  __shared__ alignas(16) __bf16 Plds[4][16][40];

  const int lane = threadIdx.x & 31, wave = threadIdx.x >> 5;
  const int half = lane >> 4, l16 = lane & 15;
  const int b = blockIdx.z, h = blockIdx.y;
  const int qt = blockIdx.x * 64 + wave * 16;       // first query row of wave

  // Q fragments (A layout), rows qt..qt+15, dims h*64..h*64+63
  const __bf16* Qp = Q + ((size_t)(b * kS + qt)) * kD + h * kDH;
  FragA qf[2];
#pragma unroll
  for (int c = 0; c < 2; ++c) {
    const __bf16* p = Qp + (size_t)l16 * kD + c * 32;
    qf[c].h[0] = *(const bf16x8*)(p + half * 8);
    qf[c].h[1] = *(const bf16x8*)(p + 16 + half * 8);
  }

  float m_i[8], l_i[8];
  v8f acc_o[4];
#pragma unroll
  for (int v = 0; v < 8; ++v) { m_i[v] = -1e30f; l_i[v] = 0.f; }
#pragma unroll
  for (int no = 0; no < 4; ++no)
#pragma unroll
    for (int v = 0; v < 8; ++v) acc_o[no][v] = 0.f;

  const float scale = 0.125f;                        // 1/sqrt(64)
  const __bf16* Kb = Km + ((size_t)b * kS) * kD + h * kDH;
  const __bf16* Vb = Vt + ((size_t)(b * kH + h)) * kDH * kS;

  for (int j = 0; j <= qt + 15; j += 32) {           // causal key blocks
    // K^T fragments (B layout): N = key, K = dim chunk
    FragA kf[2][2];
#pragma unroll
    for (int nt = 0; nt < 2; ++nt)
#pragma unroll
      for (int c = 0; c < 2; ++c) {
        const __bf16* p = Kb + (size_t)(j + nt * 16 + l16) * kD + c * 32 + half * 8;
        kf[nt][c].h[0] = *(const bf16x8*)p;
        kf[nt][c].h[1] = *(const bf16x8*)(p + 16);
      }

    v8f s[2];
#pragma unroll
    for (int nt = 0; nt < 2; ++nt) {
#pragma unroll
      for (int v = 0; v < 8; ++v) s[nt][v] = 0.f;
      s[nt] = __builtin_amdgcn_wmma_f32_16x16x32_bf16(
          false, qf[0].v, false, kf[nt][0].v, (short)0, s[nt], false, false);
      s[nt] = __builtin_amdgcn_wmma_f32_16x16x32_bf16(
          false, qf[1].v, false, kf[nt][1].v, (short)0, s[nt], false, false);
    }

    // scale + causal mask (C layout: row = v + half*8, col = nt*16 + l16)
    float sv[2][8];
    if (j + 31 > qt) {                               // diagonal block: mask
#pragma unroll
      for (int nt = 0; nt < 2; ++nt)
#pragma unroll
        for (int v = 0; v < 8; ++v) {
          int qg = qt + v + half * 8;
          int kg = j + nt * 16 + l16;
          float val = s[nt][v] * scale;
          sv[nt][v] = (kg <= qg) ? val : -1e30f;
        }
    } else {                                         // fully-visible block
#pragma unroll
      for (int nt = 0; nt < 2; ++nt)
#pragma unroll
        for (int v = 0; v < 8; ++v) sv[nt][v] = s[nt][v] * scale;
    }

    // online softmax; row reduction stays within each 16-lane half (xor<16)
    float alpha[8], pnew[2][8];
#pragma unroll
    for (int v = 0; v < 8; ++v) {
      float x = fmaxf(sv[0][v], sv[1][v]);
      x = fmaxf(x, __shfl_xor(x, 1));
      x = fmaxf(x, __shfl_xor(x, 2));
      x = fmaxf(x, __shfl_xor(x, 4));
      x = fmaxf(x, __shfl_xor(x, 8));
      float mnew = fmaxf(m_i[v], x);
      alpha[v] = __expf(m_i[v] - mnew);
      m_i[v] = mnew;
      float p0 = __expf(sv[0][v] - mnew);
      float p1 = __expf(sv[1][v] - mnew);
      pnew[0][v] = p0; pnew[1][v] = p1;
      float rs = p0 + p1;
      rs += __shfl_xor(rs, 1);
      rs += __shfl_xor(rs, 2);
      rs += __shfl_xor(rs, 4);
      rs += __shfl_xor(rs, 8);
      l_i[v] = l_i[v] * alpha[v] + rs;
    }
#pragma unroll
    for (int no = 0; no < 4; ++no)
#pragma unroll
      for (int v = 0; v < 8; ++v) acc_o[no][v] *= alpha[v];

    // P: C layout -> LDS -> A layout (same-wave LDS is in-order; explicit wait)
#pragma unroll
    for (int nt = 0; nt < 2; ++nt)
#pragma unroll
      for (int v = 0; v < 8; ++v)
        Plds[wave][v + half * 8][nt * 16 + l16] = (__bf16)pnew[nt][v];
    asm volatile("s_wait_dscnt 0" ::: "memory");
    FragA pf;
    pf.h[0] = *(const bf16x8*)&Plds[wave][l16][half * 8];
    pf.h[1] = *(const bf16x8*)&Plds[wave][l16][16 + half * 8];

    // O += P * V  (V^T fragments: N = dim, K = key; Vt is [bh][dim][s])
#pragma unroll
    for (int no = 0; no < 4; ++no) {
      const __bf16* p = Vb + (size_t)(no * 16 + l16) * kS + j + half * 8;
      FragA vf;
      vf.h[0] = *(const bf16x8*)p;
      vf.h[1] = *(const bf16x8*)(p + 16);
      acc_o[no] = __builtin_amdgcn_wmma_f32_16x16x32_bf16(
          false, pf.v, false, vf.v, (short)0, acc_o[no], false, false);
    }
  }

  // normalize and store ctx (bf16), interleaved head layout [bs, h*64+dim]
#pragma unroll
  for (int no = 0; no < 4; ++no)
#pragma unroll
    for (int v = 0; v < 8; ++v) {
      float o = acc_o[no][v] / l_i[v];
      int row = qt + v + half * 8;
      ctx[((size_t)(b * kS + row)) * kD + h * kDH + no * 16 + l16] = (__bf16)o;
    }
}

// ---------------------------------------------------------------------------
// Host-side orchestration
// ---------------------------------------------------------------------------
extern "C" void kernel_launch(void* const* d_in, const int* in_sizes, int n_in,
                              void* d_out, int out_size, void* d_ws, size_t ws_size,
                              hipStream_t stream) {
  (void)in_sizes; (void)n_in; (void)out_size; (void)ws_size;

  const float* x   = (const float*)d_in[0];
  const float* fqw = (const float*)d_in[1];
  const float* fkw = (const float*)d_in[2];
  const float* fvw = (const float*)d_in[3];
  const float* rqw = (const float*)d_in[4];
  const float* rkw = (const float*)d_in[5];
  const float* rvw = (const float*)d_in[6];
  const float* fqn = (const float*)d_in[7];
  const float* fkn = (const float*)d_in[8];
  const float* fvn = (const float*)d_in[9];
  const float* rqn = (const float*)d_in[10];
  const float* rkn = (const float*)d_in[11];
  const float* rvn = (const float*)d_in[12];
  const float* Wo  = (const float*)d_in[13];

  char* ws = (char*)d_ws;
  size_t off = 0;
  auto alloc = [&](size_t bytes) -> char* {
    char* p = ws + off;
    off += (bytes + 255) & ~(size_t)255;
    return p;
  };

  __bf16* xbf = (__bf16*)alloc((size_t)kTOK * kD * 2);
  __bf16* fqB = (__bf16*)alloc((size_t)kNR * kD * 2);
  __bf16* fkB = (__bf16*)alloc((size_t)kNR * kD * 2);
  __bf16* fvB = (__bf16*)alloc((size_t)kNR * kD * 2);
  __bf16* rqB = (__bf16*)alloc((size_t)kD * kNR * 2);
  __bf16* rkB = (__bf16*)alloc((size_t)kD * kNR * 2);
  __bf16* rvB = (__bf16*)alloc((size_t)kD * kNR * 2);
  __bf16* WoB = (__bf16*)alloc((size_t)kD * kD * 2);
  float*  hq  = (float*)alloc((size_t)kTOK * kR * 4);
  float*  hk  = (float*)alloc((size_t)kTOK * kR * 4);
  float*  hv  = (float*)alloc((size_t)kTOK * kR * 4);
  __bf16* Qbf = (__bf16*)alloc((size_t)kTOK * kD * 2);
  __bf16* Kbf = (__bf16*)alloc((size_t)kTOK * kD * 2);
  __bf16* Vbf = (__bf16*)alloc((size_t)kTOK * kD * 2);
  char*   SCR = alloc((size_t)kTOK * kNR * 4);      // 32 MB reused scratch

  float*  xn   = (float*)SCR;                        // [4096, 2048] f32
  __bf16* z    = (__bf16*)SCR;                       // [4096, 2048] bf16 (later)
  __bf16* Vt   = (__bf16*)SCR;                       // [B*H, 64, S] bf16 (later)
  __bf16* ctxb = (__bf16*)(SCR + (size_t)kTOK * kD * 2);  // [4096, 1024] bf16

  // ---- conversions ----
  f32_to_bf16_kernel<<<(kTOK * kD) / 256, 256, 0, stream>>>(x, xbf, kTOK * kD);
  compressB_kernel<<<(kNR * kD) / 256, 256, 0, stream>>>(fqn, fqB);
  compressB_kernel<<<(kNR * kD) / 256, 256, 0, stream>>>(fkn, fkB);
  compressB_kernel<<<(kNR * kD) / 256, 256, 0, stream>>>(fvn, fvB);
  restoreB_kernel<<<(kD * kNR) / 256, 256, 0, stream>>>(rqn, rqB);
  restoreB_kernel<<<(kD * kNR) / 256, 256, 0, stream>>>(rkn, rkB);
  restoreB_kernel<<<(kD * kNR) / 256, 256, 0, stream>>>(rvn, rvB);
  f32_to_bf16_kernel<<<(kD * kD) / 256, 256, 0, stream>>>(Wo, WoB, kD * kD);

  dim3 gemm_blk(256);
  dim3 grid_c(kNR / 256, kTOK / 128);  // compress: N=2048
  dim3 grid_r(kD / 256, kTOK / 128);   // restore/out: N=1024

  // ---- compress: xn = x * neurons, h = w-weighted reduce over N ----
  gemm_bf16_kernel<false><<<grid_c, gemm_blk, 0, stream>>>(xbf, fqB, xn, kTOK, kNR, kD);
  compress_reduce_kernel<<<kTOK, kR, 0, stream>>>(xn, fqw, hq);
  gemm_bf16_kernel<false><<<grid_c, gemm_blk, 0, stream>>>(xbf, fkB, xn, kTOK, kNR, kD);
  compress_reduce_kernel<<<kTOK, kR, 0, stream>>>(xn, fkw, hk);
  gemm_bf16_kernel<false><<<grid_c, gemm_blk, 0, stream>>>(xbf, fvB, xn, kTOK, kNR, kD);
  compress_reduce_kernel<<<kTOK, kR, 0, stream>>>(xn, fvw, hv);

  // ---- restore: z = w (x) h, then Q/K/V = z * neuronsR ----
  build_z_kernel<<<kTOK, 256, 0, stream>>>(rqw, hq, z);
  gemm_bf16_kernel<true><<<grid_r, gemm_blk, 0, stream>>>(z, rqB, Qbf, kTOK, kD, kNR);
  build_z_kernel<<<kTOK, 256, 0, stream>>>(rkw, hk, z);
  gemm_bf16_kernel<true><<<grid_r, gemm_blk, 0, stream>>>(z, rkB, Kbf, kTOK, kD, kNR);
  build_z_kernel<<<kTOK, 256, 0, stream>>>(rvw, hv, z);
  gemm_bf16_kernel<true><<<grid_r, gemm_blk, 0, stream>>>(z, rvB, Vbf, kTOK, kD, kNR);

  // ---- attention ----
  transposeV_kernel<<<(kB * kH * kDH * kS) / 256, 256, 0, stream>>>(Vbf, Vt);
  flash_attn_kernel<<<dim3(kS / 64, kH, kB), 128, 0, stream>>>(Qbf, Kbf, Vt, ctxb);

  // ---- output projection: out = ctx * W_o^T (Bt == W_o row-major) ----
  gemm_bf16_kernel<false><<<grid_r, gemm_blk, 0, stream>>>(ctxb, WoB, (float*)d_out,
                                                           kTOK, kD, kD);
}
